// MultiHeadAttention_66486093742402
// MI455X (gfx1250) — matrix operand order repaired and checked by reading, not compiled, not run on hardware
//
#include <hip/hip_runtime.h>

typedef __attribute__((ext_vector_type(16))) _Float16 v16h;
typedef __attribute__((ext_vector_type(8)))  _Float16 v8h;
typedef __attribute__((ext_vector_type(8)))  float    v8f;

#define DIM   1024
#define HEADS 16
#define DEPTH 64
#define SEQ   2048
#define BATCH 4
#define SCS   2052   // padded LDS row stride in floats (8*SCS % 64 != 0 -> conflict-free halves)

// ---------------- WMMA fragment loaders ----------------
// A (16x32 f16): lane L -> row m=L&15, hi=L>>4; holds K = [hi*8 .. +7] and [16+hi*8 .. +7]
__device__ __forceinline__ v16h load_a_f32(const float* __restrict__ base, long row_stride,
                                           int m, int hi, int kbase) {
    const float* p0 = base + (long)m * row_stride + kbase + hi * 8;
    v16h a;
#pragma unroll
    for (int i = 0; i < 8; ++i) a[i] = (_Float16)p0[i];
#pragma unroll
    for (int i = 0; i < 8; ++i) a[8 + i] = (_Float16)p0[16 + i];
    return a;
}

__device__ __forceinline__ v16h load_a_f16(const _Float16* __restrict__ base, long row_stride,
                                           int m, int hi, int kbase) {
    const _Float16* p0 = base + (long)m * row_stride + kbase + hi * 8;
    v8h lo = *(const v8h*)p0;
    v8h hv = *(const v8h*)(p0 + 16);
    v16h a;
#pragma unroll
    for (int i = 0; i < 8; ++i) { a[i] = lo[i]; a[8 + i] = hv[i]; }
    return a;
}

// B (32x16 f16): lane L -> col n=L&15, hi=L>>4; holds K = [hi*8 .. +7] and [16+hi*8 .. +7]
__device__ __forceinline__ v16h load_b_f32(const float* __restrict__ base, long row_stride,
                                           int n, int hi, int kbase) {
    const float* p0 = base + (long)n * row_stride + kbase + hi * 8;
    v16h b;
#pragma unroll
    for (int i = 0; i < 8; ++i) b[i] = (_Float16)p0[i];
#pragma unroll
    for (int i = 0; i < 8; ++i) b[8 + i] = (_Float16)p0[16 + i];
    return b;
}

__device__ __forceinline__ v16h load_b_f16(const _Float16* __restrict__ base, long row_stride,
                                           int n, int hi, int kbase) {
    const _Float16* p0 = base + (long)n * row_stride + kbase + hi * 8;
    v8h lo = *(const v8h*)p0;   // 16B-aligned
    v8h hv = *(const v8h*)(p0 + 16);
    v16h b;
#pragma unroll
    for (int i = 0; i < 8; ++i) { b[i] = lo[i]; b[8 + i] = hv[i]; }
    return b;
}

// ---------------- Kernel 1: fused QKV projection (16x64 tile per wave) ----------------
// z=0: q@WQ^T -> qh f16 [B*S, DIM]; z=1: k@WK^T -> kh; z=2: v@WK^T (reference bug) -> vT [B,H,DEPTH,S]
// Each wave computes 4 N-subtiles sharing one A fragment per k-step (4x fewer A loads/converts).
__global__ void mha_proj_kernel(const float* __restrict__ q, const float* __restrict__ k,
                                const float* __restrict__ v, const float* __restrict__ WQ,
                                const float* __restrict__ WK,
                                _Float16* __restrict__ qh, _Float16* __restrict__ kh,
                                _Float16* __restrict__ vT) {
    const int lane = threadIdx.x & 31;
    const int nl = lane & 15, hi = lane >> 4;
    const int mt  = blockIdx.x;       // 0..511 : 16-row tile over B*S
    const int nt4 = blockIdx.y;       // 0..15  : 64-col tile over DIM
    const int which = blockIdx.z;     // 0:q 1:k 2:v
    const float* X = (which == 0) ? q : ((which == 1) ? k : v);
    const float* W = (which == 0) ? WQ : WK;
    const int row0 = mt * 16, col0 = nt4 * 64;

    v8f zero = {};
    v8f acc[4] = {zero, zero, zero, zero};
    for (int kb = 0; kb < DIM; kb += 32) {
        v16h a = load_a_f32(X + (long)row0 * DIM, DIM, nl, hi, kb);
#pragma unroll
        for (int tsub = 0; tsub < 4; ++tsub) {
            v16h b = load_b_f32(W + (long)(col0 + tsub * 16) * DIM, DIM, nl, hi, kb);
            acc[tsub] = __builtin_amdgcn_wmma_f32_16x16x32_f16(false, a, false, b,
                                                               (short)0, acc[tsub], false, false);
        }
    }

    if (which < 2) {
        _Float16* Y = (which == 0) ? qh : kh;
#pragma unroll
        for (int tsub = 0; tsub < 4; ++tsub) {
            const int e = col0 + tsub * 16 + nl;
#pragma unroll
            for (int r = 0; r < 8; ++r) {
                int mg = row0 + hi * 8 + r;
                Y[(long)mg * DIM + e] = (_Float16)acc[tsub][r];
            }
        }
    } else {
#pragma unroll
        for (int tsub = 0; tsub < 4; ++tsub) {
            const int e = col0 + tsub * 16 + nl;
            const int h = e >> 6, dd = e & 63;
            const int mg0 = row0 + hi * 8;
            const int bb = mg0 >> 11, s0 = mg0 & (SEQ - 1);
            // 8 consecutive s values -> one contiguous 16B f16 store per lane
            v8h pk;
#pragma unroll
            for (int r = 0; r < 8; ++r) pk[r] = (_Float16)acc[tsub][r];
            *(v8h*)(vT + (((long)bb * HEADS + h) * DEPTH + dd) * SEQ + s0) = pk;
        }
    }
}

// ---------------- Kernel 2: fused scores -> softmax -> AV ----------------
// One 256-thread workgroup handles 16 query rows for one (b,h):
//   phase 1: 16x2048 scores into LDS (8 waves x 16 WMMA tiles, Q frags reused)
//   phase 2: row softmax in LDS; normalized probs written once to att (only HBM touch)
//   phase 3: out[16 x 64] = probs @ Vh, probs read from LDS, V from transposed f16 staging
__global__ void mha_fused_kernel(const _Float16* __restrict__ qh,
                                 const _Float16* __restrict__ kh,
                                 const _Float16* __restrict__ vT,
                                 float* __restrict__ att,
                                 float* __restrict__ out) {
    extern __shared__ float lds[];
    float* sc    = lds;                 // [16][SCS] scores/probs
    float* redAV = lds + 16 * SCS;      // [4][256]  AV cross-wave combine
    float* red   = redAV + 4 * 256;     // [256]     softmax reductions
    float* inv16 = red + 256;           // [16]      per-row 1/sum

    const int t = threadIdx.x;          // 0..255
    const int lane = t & 31;
    const int w = t >> 5;               // wave 0..7
    const int nl = lane & 15, hi = lane >> 4;

    const int it = blockIdx.x;          // query tile 0..127
    const int bh = blockIdx.y;          // 0..63
    const int b = bh >> 4, h = bh & 15;
    const long qrow0 = (long)b * SEQ + it * 16;
    const int colbase = h * DEPTH;

    // ---- phase 1: scores ----
    v16h a0 = load_a_f16(qh + qrow0 * DIM + colbase, DIM, nl, hi, 0);
    v16h a1 = load_a_f16(qh + qrow0 * DIM + colbase, DIM, nl, hi, 32);
#pragma unroll 4
    for (int tile = 0; tile < 16; ++tile) {
        const int j0 = w * 256 + tile * 16;                // key index base
        const long krow0 = (long)b * SEQ + j0;
        v8f acc = {};
        v16h b0 = load_b_f16(kh + krow0 * DIM + colbase, DIM, nl, hi, 0);
        acc = __builtin_amdgcn_wmma_f32_16x16x32_f16(false, a0, false, b0,
                                                     (short)0, acc, false, false);
        v16h b1 = load_b_f16(kh + krow0 * DIM + colbase, DIM, nl, hi, 32);
        acc = __builtin_amdgcn_wmma_f32_16x16x32_f16(false, a1, false, b1,
                                                     (short)0, acc, false, false);
#pragma unroll
        for (int r = 0; r < 8; ++r)
            sc[(hi * 8 + r) * SCS + j0 + nl] = acc[r] * 0.125f;   // 1/sqrt(64)
    }
    __syncthreads();

    // ---- phase 2: softmax over each of the 16 rows ----
    {
        const int row = t >> 4, l16 = t & 15;
        float* rp = sc + row * SCS;
        float vmax = -3.402823466e38f;
        for (int i = 0; i < 128; ++i) vmax = fmaxf(vmax, rp[l16 + i * 16]);
        red[t] = vmax; __syncthreads();
        for (int s = 8; s > 0; s >>= 1) {
            if (l16 < s) red[t] = fmaxf(red[t], red[t + s]);
            __syncthreads();
        }
        vmax = red[row * 16];
        float sum = 0.0f;
        for (int i = 0; i < 128; ++i) {
            float e = __expf(rp[l16 + i * 16] - vmax);
            rp[l16 + i * 16] = e;
            sum += e;
        }
        __syncthreads();                 // everyone done reading red (max phase)
        red[t] = sum; __syncthreads();
        for (int s = 8; s > 0; s >>= 1) {
            if (l16 < s) red[t] += red[t + s];
            __syncthreads();
        }
        if (l16 == 0) inv16[row] = 1.0f / red[row * 16];
    }
    __syncthreads();

    // normalize in LDS + coalesced att writeout (the only HBM traffic for probs)
    {
        const long abase = ((long)bh * SEQ + it * 16) * SEQ;
        for (int j = 0; j < 128; ++j) {
            const int flat = t + j * 256;
            const int row = flat >> 11, col = flat & (SEQ - 1);
            const float val = sc[row * SCS + col] * inv16[row];
            sc[row * SCS + col] = val;
            att[abase + flat] = val;
        }
    }
    __syncthreads();

    // ---- phase 3: out = probs @ Vh ----
    const int nt = w & 3;               // depth tile 0..3
    const int ks = w >> 2;              // K half 0..1
    const _Float16* Bp = vT + (((long)b * HEADS + h) * DEPTH + nt * 16) * SEQ;

    v8f acc = {};
    const int kb0 = ks * (SEQ / 2);
    for (int kb = kb0; kb < kb0 + SEQ / 2; kb += 32) {
        // A fragment from LDS probabilities (f32 -> f16)
        const float* p0 = sc + nl * SCS + kb + hi * 8;
        v16h a;
#pragma unroll
        for (int i = 0; i < 8; ++i) a[i] = (_Float16)p0[i];
#pragma unroll
        for (int i = 0; i < 8; ++i) a[8 + i] = (_Float16)p0[16 + i];
        v16h bb = load_b_f16(Bp, SEQ, nl, hi, kb);
        acc = __builtin_amdgcn_wmma_f32_16x16x32_f16(false, a, false, bb,
                                                     (short)0, acc, false, false);
    }

    float* rv = redAV + nt * 256;
    if (ks == 0) {
#pragma unroll
        for (int r = 0; r < 8; ++r) rv[r * 32 + lane] = acc[r];
    }
    __syncthreads();
    if (ks == 1) {
        const int e = h * DEPTH + nt * 16 + nl;
#pragma unroll
        for (int r = 0; r < 8; ++r) {
            float valf = acc[r] + rv[r * 32 + lane];
            int s = it * 16 + hi * 8 + r;
            out[((long)b * SEQ + s) * DIM + e] = valf;
        }
    }
}

// ---------------- host launcher ----------------
extern "C" void kernel_launch(void* const* d_in, const int* in_sizes, int n_in,
                              void* d_out, int out_size, void* d_ws, size_t ws_size,
                              hipStream_t stream) {
    const float* q  = (const float*)d_in[0];
    const float* k  = (const float*)d_in[1];
    const float* v  = (const float*)d_in[2];
    const float* WQ = (const float*)d_in[3];
    const float* WK = (const float*)d_in[4];
    // d_in[5] == WV intentionally unused (reference projects V with WK).

    _Float16* qh = (_Float16*)d_ws;                      // [B*S, DIM] f16
    _Float16* kh = qh + (size_t)BATCH * SEQ * DIM;       // [B*S, DIM] f16
    _Float16* vT = kh + (size_t)BATCH * SEQ * DIM;       // [B, H, DEPTH, S] f16

    float* out = (float*)d_out;                                  // [B, S, DIM]
    float* att = out + (size_t)BATCH * SEQ * DIM;                // [B, H, S, S]

    mha_proj_kernel<<<dim3(BATCH * SEQ / 16, DIM / 64, 3), dim3(32), 0, stream>>>(
        q, k, v, WQ, WK, qh, kh, vT);

    const size_t lds_bytes = (size_t)(16 * SCS + 4 * 256 + 256 + 16) * sizeof(float);
    mha_fused_kernel<<<dim3(SEQ / 16, BATCH * HEADS), dim3(256), lds_bytes, stream>>>(
        qh, kh, vT, att, out);
}
